// FlexAttn_12335146074379
// MI455X (gfx1250) — compile-verified, matrix operand order checked by reading, not run on hardware
//
#include <hip/hip_runtime.h>
#include <hip/hip_bf16.h>
#include <stdint.h>

#define BB 4
#define NN 1024
#define EE 1024
#define HH 16
#define DD 64
#define SCALE_F 0.125f
#define M_TOT (BB * NN)   // 4096

typedef __attribute__((ext_vector_type(16))) __bf16          v16bf;
typedef __attribute__((ext_vector_type(8)))  float           v8f;
typedef __attribute__((ext_vector_type(8)))  unsigned short  v8us;
typedef __attribute__((ext_vector_type(16))) unsigned short  v16us;

__device__ __forceinline__ unsigned short f2bf(float f) {
  unsigned u = __builtin_bit_cast(unsigned, f);
  u += 0x7FFFu + ((u >> 16) & 1u);      // round-to-nearest-even
  return (unsigned short)(u >> 16);
}

// Load one v16bf A/B fragment: two contiguous 16B chunks (K and K+16 halves).
__device__ __forceinline__ v16bf ld_frag(const unsigned short* p) {
  v8us a0 = *reinterpret_cast<const v8us*>(p);
  v8us a1 = *reinterpret_cast<const v8us*>(p + 16);
  v16us w = __builtin_shufflevector(a0, a1, 0,1,2,3,4,5,6,7,8,9,10,11,12,13,14,15);
  return __builtin_bit_cast(v16bf, w);
}

__device__ __forceinline__ v8f wmma_bf16(v16bf a, v16bf b, v8f c) {
  return __builtin_amdgcn_wmma_f32_16x16x32_bf16(false, a, false, b, (short)0, c,
                                                 false, false);
}

// Async global->LDS copy of 16 bytes (GVS form), tracked by ASYNCcnt.
__device__ __forceinline__ void async_b128(void* lds_ptr, const void* sbase,
                                           unsigned byte_off) {
  unsigned lds_addr = (unsigned)(size_t)lds_ptr;   // low 32b of LDS aperture addr
  asm volatile("global_load_async_to_lds_b128 %0, %1, %2"
               :: "v"(lds_addr), "v"(byte_off), "s"(sbase)
               : "memory");
}

__device__ __forceinline__ void wait_async0() {
  asm volatile("s_wait_asynccnt 0x0" ::: "memory");
}

// ---------------------------------------------------------------------------
// Prep kernels: one-time f32 -> bf16 conversion (x) and convert+transpose (W)
// ---------------------------------------------------------------------------
__global__ __launch_bounds__(256)
void convert_kernel(const float* __restrict__ src,
                    unsigned short* __restrict__ dst) {
  int i = (blockIdx.x * 256 + threadIdx.x) * 4;
  float4 v = *reinterpret_cast<const float4*>(src + i);
  uint2 pk;
  pk.x = (unsigned)f2bf(v.x) | ((unsigned)f2bf(v.y) << 16);
  pk.y = (unsigned)f2bf(v.z) | ((unsigned)f2bf(v.w) << 16);
  *reinterpret_cast<uint2*>(dst + i) = pk;
}

// W [1024][ncols] f32  ->  Wt [ncols][1024] bf16
__global__ __launch_bounds__(256)
void convert_transpose_kernel(const float* __restrict__ W,
                              unsigned short* __restrict__ Wt, int ncols) {
  __shared__ float tile[32][33];
  const int c0 = blockIdx.x * 32;
  const int k0 = blockIdx.y * 32;
  const int tx = threadIdx.x & 31;
  const int ty = threadIdx.x >> 5;   // 0..7
  #pragma unroll
  for (int i = 0; i < 4; ++i) {
    int kr = ty + i * 8;
    tile[kr][tx] = W[(size_t)(k0 + kr) * ncols + c0 + tx];
  }
  __syncthreads();
  #pragma unroll
  for (int i = 0; i < 4; ++i) {
    int cr = ty + i * 8;
    Wt[(size_t)(c0 + cr) * EE + k0 + tx] = f2bf(tile[tx][cr]);
  }
}

// ---------------------------------------------------------------------------
// Kernel 1: qkv = x @ Wqkv + bqkv. A,B already bf16 (B pre-transposed).
// Output: Q,K bf16 [B,H,N,D]; V bf16 transposed [B,H,D,N].
// ---------------------------------------------------------------------------
#define GT_M 128
#define GT_N 128
#define GT_K 32
#define LDA  40   // padded LDS stride (elems): 80B rows, 16B aligned, conflict-free

__global__ __launch_bounds__(256)
void qkv_gemm_kernel(const unsigned short* __restrict__ xb,
                     const unsigned short* __restrict__ Wt,   // [3E][E]
                     const float* __restrict__ bqkv,
                     unsigned short* __restrict__ Qg,
                     unsigned short* __restrict__ Kg,
                     unsigned short* __restrict__ Vtg)
{
  __shared__ unsigned short As[2][GT_M * LDA];
  __shared__ unsigned short Ws[2][GT_N * LDA];

  const int tid  = threadIdx.x;
  const int wave = tid >> 5;
  const int lane = tid & 31;
  const int ln   = lane & 15;
  const int kh   = lane >> 4;

  const int n0 = blockIdx.x * GT_N;
  const int m0 = blockIdx.y * GT_M;
  const int wm = wave >> 1;
  const int wn = wave & 1;

  auto issue_tile = [&](int k0, int buf) {
    #pragma unroll
    for (int it = 0; it < 2; ++it) {         // 512 slots / 256 threads
      int slot = it * 256 + tid;
      int row  = slot >> 2;
      int cs   = (slot & 3) * 8;
      async_b128(&As[buf][row * LDA + cs], xb,
                 (unsigned)(((size_t)(m0 + row) * EE + k0 + cs) * 2));
      async_b128(&Ws[buf][row * LDA + cs], Wt,
                 (unsigned)(((size_t)(n0 + row) * EE + k0 + cs) * 2));
    }
  };

  v8f acc[2][4];
  #pragma unroll
  for (int i = 0; i < 2; ++i)
    #pragma unroll
    for (int j = 0; j < 4; ++j)
      #pragma unroll
      for (int e = 0; e < 8; ++e) acc[i][j][e] = 0.0f;

  issue_tile(0, 0);
  for (int t = 0; t < EE / GT_K; ++t) {
    wait_async0();
    __syncthreads();
    if (t + 1 < EE / GT_K) issue_tile((t + 1) * GT_K, (t + 1) & 1);

    const int buf = t & 1;
    v16bf af[2], bfm[4];
    #pragma unroll
    for (int i = 0; i < 2; ++i)
      af[i] = ld_frag(&As[buf][(wm * 32 + i * 16 + ln) * LDA + kh * 8]);
    #pragma unroll
    for (int j = 0; j < 4; ++j)
      bfm[j] = ld_frag(&Ws[buf][(wn * 64 + j * 16 + ln) * LDA + kh * 8]);

    #pragma unroll
    for (int i = 0; i < 2; ++i)
      #pragma unroll
      for (int j = 0; j < 4; ++j)
        acc[i][j] = wmma_bf16(af[i], bfm[j], acc[i][j]);
  }

  // Epilogue: +bias, bf16, scatter Q/K [B,H,N,D], V transposed [B,H,D,N]
  #pragma unroll
  for (int i = 0; i < 2; ++i) {
    int gmBase = m0 + wm * 32 + i * 16 + 8 * kh;
    #pragma unroll
    for (int j = 0; j < 4; ++j) {
      int gn    = n0 + wn * 64 + j * 16 + ln;
      int which = gn >> 10;            // 0=Q 1=K 2=V
      int e     = gn & 1023;
      int h     = e >> 6;
      int d     = e & 63;
      float bias = bqkv[gn];
      #pragma unroll
      for (int r = 0; r < 8; ++r) {
        int gm = gmBase + r;
        int b  = gm >> 10;
        int n  = gm & 1023;
        unsigned short val = f2bf(acc[i][j][r] + bias);
        size_t bh = (size_t)b * HH + h;
        if (which == 0)      Qg[(bh * NN + n) * DD + d]  = val;
        else if (which == 1) Kg[(bh * NN + n) * DD + d]  = val;
        else                 Vtg[(bh * DD + d) * NN + n] = val;
      }
    }
  }
}

// ---------------------------------------------------------------------------
// Kernel 2: flash attention with additive bias. Double-buffered async K/V.
// ---------------------------------------------------------------------------
#define KSTR 72   // 144B rows, 16B aligned, conflict-free for b128 frags

__global__ __launch_bounds__(128)
void attn_kernel(const unsigned short* __restrict__ Qg,
                 const unsigned short* __restrict__ Kg,
                 const unsigned short* __restrict__ Vtg,  // [B,H,D,N]
                 const float* __restrict__ bias,
                 unsigned short* __restrict__ attn_bf)
{
  __shared__ unsigned short KsL[2][64 * KSTR];   // [key][d]
  __shared__ unsigned short VtL[2][64 * KSTR];   // [d][key]
  __shared__ unsigned short Ps[64 * KSTR];       // per-wave 16-row slices

  const int tid  = threadIdx.x;
  const int wave = tid >> 5;
  const int lane = tid & 31;
  const int ln   = lane & 15;
  const int kh   = lane >> 4;

  const int bh = blockIdx.y;          // b*H + h
  const int b  = bh >> 4;
  const int h  = bh & 15;
  const int qw = blockIdx.x * 64 + wave * 16;

  const unsigned short* Kbase  = Kg  + (size_t)bh * NN * DD;
  const unsigned short* Vtbase = Vtg + (size_t)bh * DD * NN;

  auto issue_chunk = [&](int kc, int buf) {
    #pragma unroll
    for (int it = 0; it < 4; ++it) {          // 512 slots / 128 threads
      int slot = it * 128 + tid;
      int row  = slot >> 3;                   // key row (K) / d row (Vt)
      int cs   = (slot & 7) * 8;
      async_b128(&KsL[buf][row * KSTR + cs], Kbase,
                 (unsigned)(((size_t)(kc + row) * DD + cs) * 2));
      async_b128(&VtL[buf][row * KSTR + cs], Vtbase,
                 (unsigned)(((size_t)row * NN + kc + cs) * 2));
    }
  };

  // Persistent Q A-fragments (D=64 -> two K=32 fragments)
  const unsigned short* qrow = Qg + ((size_t)bh * NN + qw + ln) * DD;
  v16bf qf0 = ld_frag(qrow + kh * 8);
  v16bf qf1 = ld_frag(qrow + 32 + kh * 8);

  const float* brow[8];
  #pragma unroll
  for (int r = 0; r < 8; ++r)
    brow[r] = bias + ((size_t)bh * NN + qw + r + 8 * kh) * NN;

  float Mrow[8], Lrow[8];
  #pragma unroll
  for (int r = 0; r < 8; ++r) { Mrow[r] = -3.0e38f; Lrow[r] = 0.0f; }

  v8f oacc[4];
  #pragma unroll
  for (int dt = 0; dt < 4; ++dt)
    #pragma unroll
    for (int e = 0; e < 8; ++e) oacc[dt][e] = 0.0f;

  issue_chunk(0, 0);
  for (int t = 0; t < NN / 64; ++t) {
    const int kc  = t * 64;
    const int buf = t & 1;
    wait_async0();
    __syncthreads();
    if (t + 1 < NN / 64) issue_chunk(kc + 64, (t + 1) & 1);

    if (kc + 64 < NN)   // pull next bias chunk toward L2
      __builtin_prefetch(brow[0] + kc + 64 + ln * 4, 0, 1);

    // S = Q K^T * scale + bias
    v8f s[4];
    #pragma unroll
    for (int nsub = 0; nsub < 4; ++nsub) {
      int key = nsub * 16 + ln;
      v16bf kb0 = ld_frag(&KsL[buf][key * KSTR + kh * 8]);
      v16bf kb1 = ld_frag(&KsL[buf][key * KSTR + 32 + kh * 8]);
      v8f c = {0.f, 0.f, 0.f, 0.f, 0.f, 0.f, 0.f, 0.f};
      c = wmma_bf16(qf0, kb0, c);
      c = wmma_bf16(qf1, kb1, c);
      #pragma unroll
      for (int r = 0; r < 8; ++r)
        s[nsub][r] = c[r] * SCALE_F + brow[r][kc + nsub * 16 + ln];
    }

    // Online softmax (row stats live in the 16-lane half-group)
    float al[8];
    #pragma unroll
    for (int r = 0; r < 8; ++r) {
      float m = fmaxf(fmaxf(s[0][r], s[1][r]), fmaxf(s[2][r], s[3][r]));
      #pragma unroll
      for (int msk = 1; msk <= 8; msk <<= 1)
        m = fmaxf(m, __shfl_xor(m, msk, 32));
      float mnew = fmaxf(Mrow[r], m);
      al[r] = __expf(Mrow[r] - mnew);
      Mrow[r] = mnew;
    }

    float ls[8];
    #pragma unroll
    for (int r = 0; r < 8; ++r) ls[r] = 0.0f;
    #pragma unroll
    for (int nsub = 0; nsub < 4; ++nsub)
      #pragma unroll
      for (int r = 0; r < 8; ++r) {
        float p = __expf(s[nsub][r] - Mrow[r]);
        ls[r] += p;
        Ps[(wave * 16 + r + 8 * kh) * KSTR + nsub * 16 + ln] = f2bf(p);
      }
    #pragma unroll
    for (int r = 0; r < 8; ++r) {
      float tt = ls[r];
      #pragma unroll
      for (int msk = 1; msk <= 8; msk <<= 1)
        tt += __shfl_xor(tt, msk, 32);
      Lrow[r] = Lrow[r] * al[r] + tt;
    }

    #pragma unroll
    for (int dt = 0; dt < 4; ++dt)
      #pragma unroll
      for (int r = 0; r < 8; ++r)
        oacc[dt][r] *= al[r];

    // O += P @ V
    const unsigned short* prow = &Ps[(wave * 16 + ln) * KSTR];
    v16bf pf0 = ld_frag(prow + kh * 8);
    v16bf pf1 = ld_frag(prow + 32 + kh * 8);
    #pragma unroll
    for (int dt = 0; dt < 4; ++dt) {
      int d = dt * 16 + ln;
      v16bf vb0 = ld_frag(&VtL[buf][d * KSTR + kh * 8]);
      v16bf vb1 = ld_frag(&VtL[buf][d * KSTR + 32 + kh * 8]);
      oacc[dt] = wmma_bf16(pf0, vb0, oacc[dt]);
      oacc[dt] = wmma_bf16(pf1, vb1, oacc[dt]);
    }
  }

  // Normalize, write bf16 [B,N,E] (feeds the out-projection GEMM directly)
  #pragma unroll
  for (int dt = 0; dt < 4; ++dt)
    #pragma unroll
    for (int r = 0; r < 8; ++r) {
      float o = oacc[dt][r] / Lrow[r];
      attn_bf[((size_t)b * NN + qw + r + 8 * kh) * EE + h * DD + dt * 16 + ln] =
          f2bf(o);
    }
}

// ---------------------------------------------------------------------------
// Kernel 3: out = attn @ Wout + bout (A bf16, B bf16 pre-transposed, out f32)
// ---------------------------------------------------------------------------
__global__ __launch_bounds__(256)
void out_gemm_kernel(const unsigned short* __restrict__ Ab,
                     const unsigned short* __restrict__ Wt,   // [E][E]
                     const float* __restrict__ bout,
                     float* __restrict__ out)
{
  __shared__ unsigned short As[2][GT_M * LDA];
  __shared__ unsigned short Ws[2][GT_N * LDA];

  const int tid  = threadIdx.x;
  const int wave = tid >> 5;
  const int lane = tid & 31;
  const int ln   = lane & 15;
  const int kh   = lane >> 4;

  const int n0 = blockIdx.x * GT_N;
  const int m0 = blockIdx.y * GT_M;
  const int wm = wave >> 1;
  const int wn = wave & 1;

  auto issue_tile = [&](int k0, int buf) {
    #pragma unroll
    for (int it = 0; it < 2; ++it) {
      int slot = it * 256 + tid;
      int row  = slot >> 2;
      int cs   = (slot & 3) * 8;
      async_b128(&As[buf][row * LDA + cs], Ab,
                 (unsigned)(((size_t)(m0 + row) * EE + k0 + cs) * 2));
      async_b128(&Ws[buf][row * LDA + cs], Wt,
                 (unsigned)(((size_t)(n0 + row) * EE + k0 + cs) * 2));
    }
  };

  v8f acc[2][4];
  #pragma unroll
  for (int i = 0; i < 2; ++i)
    #pragma unroll
    for (int j = 0; j < 4; ++j)
      #pragma unroll
      for (int e = 0; e < 8; ++e) acc[i][j][e] = 0.0f;

  issue_tile(0, 0);
  for (int t = 0; t < EE / GT_K; ++t) {
    wait_async0();
    __syncthreads();
    if (t + 1 < EE / GT_K) issue_tile((t + 1) * GT_K, (t + 1) & 1);

    const int buf = t & 1;
    v16bf af[2], bfm[4];
    #pragma unroll
    for (int i = 0; i < 2; ++i)
      af[i] = ld_frag(&As[buf][(wm * 32 + i * 16 + ln) * LDA + kh * 8]);
    #pragma unroll
    for (int j = 0; j < 4; ++j)
      bfm[j] = ld_frag(&Ws[buf][(wn * 64 + j * 16 + ln) * LDA + kh * 8]);

    #pragma unroll
    for (int i = 0; i < 2; ++i)
      #pragma unroll
      for (int j = 0; j < 4; ++j)
        acc[i][j] = wmma_bf16(af[i], bfm[j], acc[i][j]);
  }

  #pragma unroll
  for (int i = 0; i < 2; ++i) {
    int gmBase = m0 + wm * 32 + i * 16 + 8 * kh;
    #pragma unroll
    for (int j = 0; j < 4; ++j) {
      int gn = n0 + wn * 64 + j * 16 + ln;
      float bv = bout[gn];
      #pragma unroll
      for (int r = 0; r < 8; ++r)
        out[(size_t)(gmBase + r) * EE + gn] = acc[i][j][r] + bv;
    }
  }
}

// ---------------------------------------------------------------------------
extern "C" void kernel_launch(void* const* d_in, const int* in_sizes, int n_in,
                              void* d_out, int out_size, void* d_ws, size_t ws_size,
                              hipStream_t stream) {
  const float* x     = (const float*)d_in[0];
  const float* abias = (const float*)d_in[1];
  const float* Wqkv  = (const float*)d_in[2];
  const float* bqkv  = (const float*)d_in[3];
  const float* Wout  = (const float*)d_in[4];
  const float* bout  = (const float*)d_in[5];
  float* out = (float*)d_out;

  // Workspace carve-up (all bf16 as ushort)
  unsigned short* p = (unsigned short*)d_ws;
  const size_t n_x   = (size_t)M_TOT * EE;        // 4M
  const size_t n_wq  = (size_t)3 * EE * EE;       // 3M
  const size_t n_wo  = (size_t)EE * EE;           // 1M
  const size_t n_qkv = (size_t)BB * HH * NN * DD; // 4M each
  unsigned short* xb      = p;            p += n_x;
  unsigned short* Wqkv_t  = p;            p += n_wq;
  unsigned short* Wout_t  = p;            p += n_wo;
  unsigned short* Qg      = p;            p += n_qkv;
  unsigned short* Kg      = p;            p += n_qkv;
  unsigned short* Vtg     = p;            p += n_qkv;
  unsigned short* attn_bf = p;            p += n_x;

  convert_kernel<<<(int)(n_x / 4 / 256), 256, 0, stream>>>(x, xb);
  convert_transpose_kernel<<<dim3(3 * EE / 32, EE / 32), 256, 0, stream>>>(
      Wqkv, Wqkv_t, 3 * EE);
  convert_transpose_kernel<<<dim3(EE / 32, EE / 32), 256, 0, stream>>>(
      Wout, Wout_t, EE);

  qkv_gemm_kernel<<<dim3(3 * EE / GT_N, M_TOT / GT_M), 256, 0, stream>>>(
      xb, Wqkv_t, bqkv, Qg, Kg, Vtg);
  attn_kernel<<<dim3(NN / 64, BB * HH), 128, 0, stream>>>(
      Qg, Kg, Vtg, abias, attn_bf);
  out_gemm_kernel<<<dim3(EE / GT_N, M_TOT / GT_M), 256, 0, stream>>>(
      attn_bf, Wout_t, bout, out);
}
